// AudNet_82300163326024
// MI455X (gfx1250) — matrix operand order repaired and checked by reading, not compiled, not run on hardware
//
#include <hip/hip_runtime.h>
#include <stdint.h>

#define BETA   0.95f
#define THRESH 1.0f
#define NSTEPS 81
#define NFREQ  129
#define NHID   1000
#define NHID_P 1024      // NHID padded to 32
#define NLH    20
#define NOUT   10
#define BATCH  4096
#define TB     16        // batch rows per workgroup
#define K1P    160       // layer-1 K (129) padded to 32
#define BLK    512       // 16 waves
#define NWAVE  16
#define TPW    4         // 16-wide N-tiles per wave (64 tiles / 16 waves)
#define S1S    1032      // u8 spike row stride (1024 + 8 to spread LDS banks)

typedef __attribute__((ext_vector_type(16))) __bf16 v16bf;
typedef __attribute__((ext_vector_type(8)))  __bf16 v8bf;
typedef __attribute__((ext_vector_type(8)))  float  v8f;

union FragU {
  v16bf v;
  v8bf  h[2];
  unsigned short u[16];
};

// A fragment (16x32 bf16) from LDS spike bytes {0,1}, row stride S1S.
// ISA A-layout: lane<16 -> M=lane, K={kb+0..7, kb+16..23}; lane>=16 -> K shifted by 8.
__device__ __forceinline__ v16bf frag_a_u8(const unsigned char* s, int lane, int kbase) {
  const int m  = lane & 15;
  const int k0 = kbase + ((lane >> 4) << 3);
  const unsigned char* p = s + m * S1S + k0;
  uint64_t lo = *(const uint64_t*)(p);
  uint64_t hi = *(const uint64_t*)(p + 16);
  FragU f;
#pragma unroll
  for (int j = 0; j < 8; ++j) {
    f.u[j]     = ((lo >> (8 * j)) & 1ull) ? (unsigned short)0x3F80 : (unsigned short)0;
    f.u[j + 8] = ((hi >> (8 * j)) & 1ull) ? (unsigned short)0x3F80 : (unsigned short)0;
  }
  return f.v;
}

// A fragment from LDS bf16 (layer-1 inputs), row stride K1P elements.
__device__ __forceinline__ v16bf frag_a_bf16(const __bf16* s, int lane, int kbase) {
  const int m  = lane & 15;
  const int k0 = kbase + ((lane >> 4) << 3);
  FragU f;
  f.h[0] = *(const v8bf*)(s + m * K1P + k0);
  f.h[1] = *(const v8bf*)(s + m * K1P + k0 + 16);
  return f.v;
}

// B fragment (32x16 bf16) from global row-major [n][k] (leading dim ldk):
// lane = output column, K={kb+0..7, kb+16..23}, shifted by 8 for lane>=16. 16B aligned.
__device__ __forceinline__ v16bf frag_b(const __bf16* __restrict__ w, int ldk, int n0,
                                        int lane, int kbase) {
  const int n  = n0 + (lane & 15);
  const int k0 = kbase + ((lane >> 4) << 3);
  FragU f;
  f.h[0] = *(const v8bf*)(w + (size_t)n * ldk + k0);
  f.h[1] = *(const v8bf*)(w + (size_t)n * ldk + k0 + 16);
  return f.v;
}

// Convert f32 [nsrc][ksrc] -> zero-padded bf16 [ndst][kdst]
__global__ void cvt_pad(const float* __restrict__ src, __bf16* __restrict__ dst,
                        int nsrc, int ksrc, int ndst, int kdst) {
  int i = blockIdx.x * 256 + threadIdx.x;
  if (i >= ndst * kdst) return;
  int n = i / kdst, k = i - n * kdst;
  float v = (n < nsrc && k < ksrc) ? src[n * ksrc + k] : 0.0f;
  dst[i] = (__bf16)v;
}

__global__ __launch_bounds__(BLK) void snn_persistent(
    const float* __restrict__ x,
    const float* __restrict__ b1, const float* __restrict__ b2,
    const float* __restrict__ b3,
    const float* __restrict__ w4, const float* __restrict__ b4,
    const __bf16* __restrict__ w1b, const __bf16* __restrict__ w2b,
    const __bf16* __restrict__ w3b,
    float* __restrict__ out)
{
  __shared__ __align__(32) __bf16 xa[TB * K1P];          // 5120 B  (x_t tile, bf16)
  __shared__ __align__(16) unsigned char s1[TB * S1S];   // 16512 B (layer-1 spikes)
  __shared__ __align__(16) unsigned char s2[TB * S1S];   // 16512 B (layer-2 spikes)
  __shared__ float s3[TB * 32];                          // 2048 B  (layer-3 spikes, f32)

  const int tid  = threadIdx.x;
  const int lane = tid & 31;
  const int wv   = tid >> 5;
  const int row0 = blockIdx.x * TB;
  const int mrow = (lane >> 4) << 3;   // D-fragment: M = mrow + e, N = lane&15

  // ---- persistent membrane state as WMMA accumulator fragments ----
  v8f m1[TPW], m2[TPW], m3;
#pragma unroll
  for (int i = 0; i < TPW; ++i)
#pragma unroll
    for (int e = 0; e < 8; ++e) { m1[i][e] = 0.0f; m2[i][e] = 0.0f; }
#pragma unroll
  for (int e = 0; e < 8; ++e) m3[e] = 0.0f;
  float m4 = 0.0f;

  // per-lane biases (N = n0 + (lane&15), same for all 8 fragment elements)
  float bias1[TPW], bias2[TPW];
#pragma unroll
  for (int i = 0; i < TPW; ++i) {
    int n = (wv * TPW + i) * 16 + (lane & 15);
    bias1[i] = (n < NHID) ? b1[n] : 0.0f;
    bias2[i] = (n < NHID) ? b2[n] : 0.0f;
  }
  float bias3 = 0.0f;
  if (wv < 2) {
    int n = wv * 16 + (lane & 15);
    bias3 = (n < NLH) ? b3[n] : 0.0f;
  }
  const int r4 = tid / 10, o4 = tid - (tid / 10) * 10;   // layer-4 thread map

  for (int t = 0; t < NSTEPS; ++t) {
    __syncthreads();
    // ---- stage x_t tile into LDS as bf16 (x layout: [row][freq][step]) ----
    for (int i = tid; i < TB * K1P; i += BLK) {
      int r = i / K1P, k = i - r * K1P;
      float v = (k < NFREQ)
                  ? x[(size_t)(row0 + r) * (NFREQ * NSTEPS) + (size_t)k * NSTEPS + t]
                  : 0.0f;
      xa[i] = (__bf16)v;
    }
    __syncthreads();

    // =========== Layer 1: cur1 = x_t @ w1^T, fold LIF decay into accumulator ===========
#pragma unroll
    for (int i = 0; i < TPW; ++i)
#pragma unroll
      for (int e = 0; e < 8; ++e) {
        float mo = m1[i][e];
        m1[i][e] = BETA * mo - ((mo > THRESH) ? THRESH : 0.0f) + bias1[i];
      }
#pragma unroll 1
    for (int kc = 0; kc < K1P / 32; ++kc) {
      v16bf a = frag_a_bf16(xa, lane, kc * 32);
#pragma unroll
      for (int i = 0; i < TPW; ++i) {
        v16bf b = frag_b(w1b, K1P, (wv * TPW + i) * 16, lane, kc * 32);
        m1[i] = __builtin_amdgcn_wmma_f32_16x16x32_bf16(
            false, a, false, b, (short)0, m1[i], false, false);
      }
    }
#pragma unroll
    for (int i = 0; i < TPW; ++i) {
      int n = (wv * TPW + i) * 16 + (lane & 15);
#pragma unroll
      for (int e = 0; e < 8; ++e)
        s1[(mrow + e) * S1S + n] = (m1[i][e] > THRESH) ? 1 : 0;
    }
    __syncthreads();

    // =========== Layer 2: cur2 = s1 @ w2^T (dominant GEMM, w2 L2-resident) ===========
#pragma unroll
    for (int i = 0; i < TPW; ++i)
#pragma unroll
      for (int e = 0; e < 8; ++e) {
        float mo = m2[i][e];
        m2[i][e] = BETA * mo - ((mo > THRESH) ? THRESH : 0.0f) + bias2[i];
      }
#pragma unroll 1
    for (int kc = 0; kc < NHID_P / 32; ++kc) {
      v16bf a = frag_a_u8(s1, lane, kc * 32);
#pragma unroll
      for (int i = 0; i < TPW; ++i) {
        v16bf b = frag_b(w2b, NHID_P, (wv * TPW + i) * 16, lane, kc * 32);
        m2[i] = __builtin_amdgcn_wmma_f32_16x16x32_bf16(
            false, a, false, b, (short)0, m2[i], false, false);
      }
    }
#pragma unroll
    for (int i = 0; i < TPW; ++i) {
      int n = (wv * TPW + i) * 16 + (lane & 15);
#pragma unroll
      for (int e = 0; e < 8; ++e)
        s2[(mrow + e) * S1S + n] = (m2[i][e] > THRESH) ? 1 : 0;
    }
    __syncthreads();

    // =========== Layer 3: 1000 -> 20 (padded to 32), waves 0/1 only ===========
    if (wv < 2) {
      v8f c;
#pragma unroll
      for (int e = 0; e < 8; ++e) {
        float mo = m3[e];
        c[e] = BETA * mo - ((mo > THRESH) ? THRESH : 0.0f) + bias3;
      }
#pragma unroll 1
      for (int kc = 0; kc < NHID_P / 32; ++kc) {
        v16bf a = frag_a_u8(s2, lane, kc * 32);
        v16bf b = frag_b(w3b, NHID_P, wv * 16, lane, kc * 32);
        c = __builtin_amdgcn_wmma_f32_16x16x32_bf16(
            false, a, false, b, (short)0, c, false, false);
      }
      m3 = c;
      int n = wv * 16 + (lane & 15);
      if (n < NLH) {
#pragma unroll
        for (int e = 0; e < 8; ++e)
          s3[(mrow + e) * 32 + n] = (c[e] > THRESH) ? 1.0f : 0.0f;
      }
    }
    __syncthreads();

    // =========== Layer 4: 20 -> 10, scalar VALU on 160 threads (waves 0..4) ===========
    if (tid < 160) {
      float cur = b4[o4];
#pragma unroll
      for (int k = 0; k < NLH; ++k)
        cur += s3[r4 * 32 + k] * w4[o4 * NLH + k];
      float mo = m4;
      m4 = BETA * mo + cur - ((mo > THRESH) ? THRESH : 0.0f);
      float sp = (m4 > THRESH) ? 1.0f : 0.0f;
      size_t base = (size_t)t * (BATCH * NOUT) + (size_t)(row0 + r4) * NOUT + o4;
      out[base] = sp;                                        // spk_rec
      out[(size_t)NSTEPS * BATCH * NOUT + base] = m4;        // mem_rec
    }
  }
}

extern "C" void kernel_launch(void* const* d_in, const int* in_sizes, int n_in,
                              void* d_out, int out_size, void* d_ws, size_t ws_size,
                              hipStream_t stream) {
  (void)in_sizes; (void)n_in; (void)out_size; (void)ws_size;
  const float* x  = (const float*)d_in[0];
  const float* w1 = (const float*)d_in[1];
  const float* b1 = (const float*)d_in[2];
  const float* w2 = (const float*)d_in[3];
  const float* b2 = (const float*)d_in[4];
  const float* w3 = (const float*)d_in[5];
  const float* b3 = (const float*)d_in[6];
  const float* w4 = (const float*)d_in[7];
  const float* b4 = (const float*)d_in[8];
  float* out = (float*)d_out;

  // Workspace layout (bf16, zero-padded): w2b [1024][1024], w1b [1024][160], w3b [32][1024]
  __bf16* w2b = (__bf16*)d_ws;
  __bf16* w1b = (__bf16*)((char*)d_ws + (size_t)NHID_P * NHID_P * 2);
  __bf16* w3b = (__bf16*)((char*)d_ws + (size_t)NHID_P * NHID_P * 2 + (size_t)NHID_P * K1P * 2);

  cvt_pad<<<(NHID_P * NHID_P + 255) / 256, 256, 0, stream>>>(w2, w2b, NHID, NHID, NHID_P, NHID_P);
  cvt_pad<<<(NHID_P * K1P + 255) / 256, 256, 0, stream>>>(w1, w1b, NHID, NFREQ, NHID_P, K1P);
  cvt_pad<<<(32 * NHID_P + 255) / 256, 256, 0, stream>>>(w3, w3b, NLH, NHID, 32, NHID_P);

  snn_persistent<<<BATCH / TB, BLK, 0, stream>>>(x, b1, b2, b3, w4, b4, w1b, w2b, w3b, out);
}
